// CapsuleLayer_66340064854605
// MI455X (gfx1250) — matrix operand order repaired and checked by reading, not compiled, not run on hardware
//
#include <hip/hip_runtime.h>

#define B_SZ   64
#define N_CAPS 1152
#define DI     16
#define C_CAPS 32
#define DC     32
#define NPB    8          // n-values processed per block
#define CAP_EPS 1e-7f

typedef __attribute__((ext_vector_type(16))) __bf16 v16bf;
typedef __attribute__((ext_vector_type(8)))  float  v8f;

union BF16x16 { v16bf v; uint4 q[2]; };

__device__ __forceinline__ unsigned short f2bf(float f) {
    unsigned int x = __float_as_uint(f);
    x += 0x7FFFu + ((x >> 16) & 1u);          // round-to-nearest-even
    return (unsigned short)(x >> 16);
}

// ---------------------------------------------------------------- utilities
__global__ void zero_kernel(float* __restrict__ p, int n) {
    int i = blockIdx.x * blockDim.x + threadIdx.x;
    if (i < n) p[i] = 0.f;
}

__global__ void cvt_bf16_kernel(const float* __restrict__ src,
                                unsigned short* __restrict__ dst, size_t n) {
    size_t i = ((size_t)blockIdx.x * blockDim.x + threadIdx.x) * 4;
    if (i + 3 < n) {
        float4 f = *(const float4*)(src + i);
        ushort4 o;
        o.x = f2bf(f.x); o.y = f2bf(f.y); o.z = f2bf(f.z); o.w = f2bf(f.w);
        *(ushort4*)(dst + i) = o;
    }
}

// A operand: W[c,n, jt*16 + row, 0:16] as 16x32 bf16 tile, K=16..31 zeroed.
// Per ISA layout: lanes 0-15 hold K=0..7 in elems 0..7, lanes 16-31 hold K=8..15.
__device__ __forceinline__ v16bf load_a_tile(const unsigned short* __restrict__ Wb,
                                             int c, int n, int jt, int lane) {
    BF16x16 u;
    u.q[1].x = 0u; u.q[1].y = 0u; u.q[1].z = 0u; u.q[1].w = 0u;
    int half = lane >> 4;                 // which 8 of the 16 Di values
    int row  = jt * 16 + (lane & 15);     // j-row of this tile
    size_t off = (((size_t)c * N_CAPS + n) * DC + row) * DI + (size_t)half * 8;
    u.q[0] = *(const uint4*)(Wb + off);   // 8 bf16 = 16B aligned
    return u.v;
}

// ---------------------------------------------------------------- routing
// grid = (N/NPB, B/16), block = 256 (8 waves). Wave w owns capsules {w, w+8, w+16, w+24},
// both j-tiles, batch columns bbase..bbase+15 (WMMA N dim).
__global__ __launch_bounds__(256)
void routing_kernel(const unsigned short* __restrict__ Xb,
                    const unsigned short* __restrict__ Wb,
                    const float* __restrict__ Ocum,
                    float* __restrict__ S) {
    __shared__ float lds_l[C_CAPS * 16];
    __shared__ float lds_max[16];
    __shared__ float lds_inv[16];

    const int tid   = threadIdx.x;
    const int lane  = tid & 31;
    const int wv    = tid >> 5;           // 0..7
    const int bcol  = lane & 15;          // batch column within tile
    const int hi    = lane >> 4;          // j-half within C/D layout
    const int nbase = blockIdx.x * NPB;
    const int bbase = blockIdx.y * 16;
    const int b     = bbase + bcol;

    // Cumulative outputs for logit recomputation (constant over the n-loop)
    float oc[4][2][8];
    float acc[4][2][8];
#pragma unroll
    for (int cc = 0; cc < 4; ++cc) {
        const int c = wv + (cc << 3);
#pragma unroll
        for (int jt = 0; jt < 2; ++jt)
#pragma unroll
            for (int r = 0; r < 8; ++r) {
                const int j = jt * 16 + hi * 8 + r;
                oc[cc][jt][r]  = Ocum[((size_t)b * C_CAPS + c) * DC + j];
                acc[cc][jt][r] = 0.f;
            }
    }

    for (int nn = 0; nn < NPB; ++nn) {
        const int n = nbase + nn;
        if (nn + 1 < NPB) {               // pull next n's W tile toward L0/L2
            size_t poff = (((size_t)wv * N_CAPS + (n + 1)) * DC) * DI;
            __builtin_prefetch(Wb + poff, 0, 0);
        }

        // B operand: inputs[b, n, 8h:8h+8] in elems 0..7 of lane half h (K=0..15); elems 8..15 = 0 (pad K)
        BF16x16 xb;
        xb.q[0].x = 0u; xb.q[0].y = 0u; xb.q[0].z = 0u; xb.q[0].w = 0u;
        xb.q[1] = xb.q[0];
        {
            const uint4* p = (const uint4*)(Xb + ((size_t)(bbase + (lane & 15)) * N_CAPS + n) * DI);
            xb.q[0] = p[lane >> 4];
        }

        // ---- pass 1: logits l[c,b] = Ocum[b,c,:] . u_hat[c,n,:,b]
#pragma unroll
        for (int cc = 0; cc < 4; ++cc) {
            const int c = wv + (cc << 3);
            float pl = 0.f;
#pragma unroll
            for (int jt = 0; jt < 2; ++jt) {
                v16bf a = load_a_tile(Wb, c, n, jt, lane);
                v8f z = {};
                v8f uh = __builtin_amdgcn_wmma_f32_16x16x32_bf16(
                    false, a, false, xb.v, (short)0, z, false, false);
#pragma unroll
                for (int r = 0; r < 8; ++r) pl += oc[cc][jt][r] * uh[r];
            }
            pl += __shfl_xor(pl, 16, 32);            // combine the two j-halves
            if (lane < 16) lds_l[c * 16 + bcol] = pl;
        }
        __syncthreads();

        // ---- softmax over the 32 capsules (per batch column)
        if (tid < 16) {
            float m = -3.4e38f;
            for (int c = 0; c < C_CAPS; ++c) m = fmaxf(m, lds_l[c * 16 + tid]);
            float ssum = 0.f;
            for (int c = 0; c < C_CAPS; ++c) ssum += __expf(lds_l[c * 16 + tid] - m);
            lds_max[tid] = m;
            lds_inv[tid] = 1.f / ssum;
        }
        __syncthreads();

        float coef[4];
#pragma unroll
        for (int cc = 0; cc < 4; ++cc) {
            const int c = wv + (cc << 3);
            coef[cc] = __expf(lds_l[c * 16 + bcol] - lds_max[bcol]) * lds_inv[bcol];
        }

        // ---- pass 2: recompute u_hat tile, accumulate s += coef * u_hat
#pragma unroll
        for (int cc = 0; cc < 4; ++cc) {
            const int c = wv + (cc << 3);
#pragma unroll
            for (int jt = 0; jt < 2; ++jt) {
                v16bf a = load_a_tile(Wb, c, n, jt, lane);
                v8f z = {};
                v8f uh = __builtin_amdgcn_wmma_f32_16x16x32_bf16(
                    false, a, false, xb.v, (short)0, z, false, false);
#pragma unroll
                for (int r = 0; r < 8; ++r) acc[cc][jt][r] += coef[cc] * uh[r];
            }
        }
        __syncthreads();                  // lds_l reused next n
    }

    // partial s reduction across n-blocks
#pragma unroll
    for (int cc = 0; cc < 4; ++cc) {
        const int c = wv + (cc << 3);
#pragma unroll
        for (int jt = 0; jt < 2; ++jt)
#pragma unroll
            for (int r = 0; r < 8; ++r) {
                const int j = jt * 16 + hi * 8 + r;
                atomicAdd(&S[((size_t)b * C_CAPS + c) * DC + j], acc[cc][jt][r]);
            }
    }
}

// ---------------------------------------------------------------- squash
// One wave per (b,c): out = s * |s|^2/(1+|s|^2)/sqrt(|s|^2+eps); Ocum += out;
// re-zero s for the next iteration; final iteration writes d_out.
__global__ __launch_bounds__(32)
void squash_kernel(float* __restrict__ S, float* __restrict__ Ocum,
                   float* __restrict__ out, int last) {
    const int bc = blockIdx.x;
    const int j  = threadIdx.x;
    const size_t idx = (size_t)bc * DC + j;
    float v  = S[idx];
    float sq = v * v;
#pragma unroll
    for (int o = 16; o > 0; o >>= 1) sq += __shfl_xor(sq, o, 32);
    float scale = sq / (1.f + sq) * rsqrtf(sq + CAP_EPS);
    float o = scale * v;
    Ocum[idx] += o;
    if (last) out[idx] = o;
    S[idx] = 0.f;
}

// ---------------------------------------------------------------- launch
extern "C" void kernel_launch(void* const* d_in, const int* in_sizes, int n_in,
                              void* d_out, int out_size, void* d_ws, size_t ws_size,
                              hipStream_t stream) {
    (void)in_sizes; (void)n_in; (void)out_size; (void)ws_size;
    const float* X = (const float*)d_in[0];    // [B, N, Di]
    const float* W = (const float*)d_in[1];    // [C, N, Dc, Di]
    float* out = (float*)d_out;                // [B, C, Dc]

    char* ws = (char*)d_ws;
    const size_t nW = (size_t)C_CAPS * N_CAPS * DC * DI;   // 18,874,368
    const size_t nX = (size_t)B_SZ * N_CAPS * DI;          // 1,179,648
    const size_t wbBytes = (2 * nW + 255) & ~(size_t)255;
    const size_t xbBytes = (2 * nX + 255) & ~(size_t)255;

    unsigned short* Wb = (unsigned short*)ws;
    unsigned short* Xb = (unsigned short*)(ws + wbBytes);
    float* S    = (float*)(ws + wbBytes + xbBytes);
    float* Ocum = S + (size_t)B_SZ * C_CAPS * DC;

    // zero s and Ocum (adjacent)
    const int nzero = 2 * B_SZ * C_CAPS * DC;
    zero_kernel<<<(nzero + 255) / 256, 256, 0, stream>>>(S, nzero);

    // fp32 -> bf16 conversion (W stays L2-resident: 37.7 MB < 192 MB)
    cvt_bf16_kernel<<<(int)((nW / 4 + 255) / 256), 256, 0, stream>>>(W, Wb, nW);
    cvt_bf16_kernel<<<(int)((nX / 4 + 255) / 256), 256, 0, stream>>>(X, Xb, nX);

    dim3 rgrid(N_CAPS / NPB, B_SZ / 16);
    for (int it = 0; it < 3; ++it) {
        routing_kernel<<<rgrid, 256, 0, stream>>>(Xb, Wb, Ocum, S);
        squash_kernel<<<B_SZ * C_CAPS, 32, 0, stream>>>(S, Ocum, out, it == 2);
    }
}
